// GPT2SelfAttention_4088808866322
// MI455X (gfx1250) — matrix operand order repaired and checked by reading, not compiled, not run on hardware
//
#include <hip/hip_runtime.h>
#include <hip/hip_bf16.h>

// GPT-2 self-attention: B=4, S=2048, E=1024, H=16, D=64, fp32 in/out.
// d_out = [out (B,S,E) | attn_weights (B,H,S,S)] fp32.
// GEMMs: v_wmma_f32_16x16x32_f16 (wave32 CDNA5), f32 accumulate.
// f16 tile staging uses global_load_async_to_lds_b128 where available.
// All attention-matrix traffic (1 GiB >> 192MB L2) is non-temporal.

typedef __attribute__((ext_vector_type(16))) _Float16 v16h;
typedef __attribute__((ext_vector_type(8)))  _Float16 v8h;
typedef __attribute__((ext_vector_type(4)))  _Float16 v4h;
typedef __attribute__((ext_vector_type(8)))  float    v8f;
typedef __attribute__((ext_vector_type(4)))  float    v4f;
typedef __attribute__((ext_vector_type(4)))  int      v4i;

#define DEV static __device__ __forceinline__
#define MASK_VALUE (-10000.0f)

// ---- async global->LDS copy of 16 bytes (falls back to vector copy) --------
// Builtin signature (per hipcc diagnostic): (AS1 int4*, AS3 int4*, Ii, Ii).
DEV void cp16_g2l(const _Float16* g, _Float16* l) {
#if defined(__gfx1250__) && __has_builtin(__builtin_amdgcn_global_load_async_to_lds_b128)
    __builtin_amdgcn_global_load_async_to_lds_b128(
        (__attribute__((address_space(1))) v4i*)g,
        (__attribute__((address_space(3))) v4i*)l, 0, 0);
#else
    *(v8h*)l = *(const v8h*)g;
#endif
}

DEV void async_join() {
#if defined(__gfx1250__) && __has_builtin(__builtin_amdgcn_s_wait_asynccnt)
    __builtin_amdgcn_s_wait_asynccnt(0);
#elif defined(__AMDGCN__)
    asm volatile("s_wait_asynccnt 0" ::: "memory");
#endif
}

// ---- WMMA fragment loaders (CDNA5 ISA 7.12.2 layouts) ----------------------
// A 16x32 f16: lane m=lane&15, g=lane>>4; f[0..7]=K(8g..8g+7), f[8..15]=K(16+8g..23+8g)
DEV v16h frag_a(const _Float16* base, int ld) {
    const int lane = threadIdx.x & 31;
    const int m = lane & 15, g = lane >> 4;
    const _Float16* p = base + m * ld + g * 8;
    v8h lo = *(const v8h*)(p);
    v8h hi = *(const v8h*)(p + 16);
    return __builtin_shufflevector(lo, hi, 0,1,2,3,4,5,6,7,8,9,10,11,12,13,14,15);
}
// B 32x16 f16 from LDS [N][K]: lane n=lane&15; f[0..7]=K(8g..8g+7), f[8..15]=K(16+8g..23+8g)
DEV v16h frag_b(const _Float16* base, int ld) {
    const int lane = threadIdx.x & 31;
    const int n = lane & 15, g = lane >> 4;
    const _Float16* p = base + n * ld + g * 8;
    v8h lo = *(const v8h*)(p);
    v8h hi = *(const v8h*)(p + 16);
    return __builtin_shufflevector(lo, hi, 0,1,2,3,4,5,6,7,8,9,10,11,12,13,14,15);
}
DEV v8f wmma(v16h a, v16h b, v8f c) {
    return __builtin_amdgcn_wmma_f32_16x16x32_f16(false, a, false, b, (short)0, c,
                                                  false, false);
}
// C/D 16x16 f32: element r -> m = r + 8*(lane>>4), n = lane&15.

DEV v4h cvt4(v4f f) {
    v4h h;
    h[0] = (_Float16)f[0]; h[1] = (_Float16)f[1];
    h[2] = (_Float16)f[2]; h[3] = (_Float16)f[3];
    return h;
}

// ---------------------------------------------------------------------------
// Kernel 1: qkv = x @ w_qkv + b_qkv ; split -> Q,K,V f16 [B,H,S,64]; Q *= 1/8.
// M=8192, K=1024, N=3072. Block 128x128, 8 waves (2x4), wave 64x32.
// ---------------------------------------------------------------------------
__global__ __launch_bounds__(256)
void qkv_gemm(const float* __restrict__ x, const float* __restrict__ w,
              const float* __restrict__ bias,
              _Float16* __restrict__ Q, _Float16* __restrict__ K,
              _Float16* __restrict__ V) {
    __shared__ __align__(16) _Float16 sA[128 * 40];
    __shared__ __align__(16) _Float16 sB[128 * 40];
    const int tid = threadIdx.x;
    const int m0 = blockIdx.y * 128, n0 = blockIdx.x * 128;
    const int wave = tid >> 5, lane = tid & 31;
    const int wm = wave >> 2, wn = wave & 3;

    v8f acc[4][2];
    #pragma unroll
    for (int i = 0; i < 4; ++i)
        #pragma unroll
        for (int j = 0; j < 2; ++j) acc[i][j] = (v8f){0,0,0,0,0,0,0,0};

    for (int k0 = 0; k0 < 1024; k0 += 32) {
        v4f ta[4], tb[4];
        #pragma unroll
        for (int it = 0; it < 4; ++it) {               // batch A loads (1024 v4f)
            int idx = tid + it * 256;
            int r = idx >> 3, c4 = (idx & 7) * 4;
            ta[it] = *(const v4f*)(x + (size_t)(m0 + r) * 1024 + k0 + c4);
        }
        #pragma unroll
        for (int it = 0; it < 4; ++it) {               // batch B loads
            int idx = tid + it * 256;
            int kk = idx >> 5, nn4 = (idx & 31) * 4;
            tb[it] = *(const v4f*)(w + (size_t)(k0 + kk) * 3072 + n0 + nn4);
        }
        #pragma unroll
        for (int it = 0; it < 4; ++it) {               // convert+store A
            int idx = tid + it * 256;
            int r = idx >> 3, c4 = (idx & 7) * 4;
            *(v4h*)(sA + r * 40 + c4) = cvt4(ta[it]);
        }
        #pragma unroll
        for (int it = 0; it < 4; ++it) {               // convert+scatter B -> [N][K]
            int idx = tid + it * 256;
            int kk = idx >> 5, nn4 = (idx & 31) * 4;
            v4h h = cvt4(tb[it]);
            #pragma unroll
            for (int j = 0; j < 4; ++j) sB[(nn4 + j) * 40 + kk] = h[j];
        }
        __syncthreads();
        v16h b0 = frag_b(sB + (wn * 32 + 0)  * 40, 40);
        v16h b1 = frag_b(sB + (wn * 32 + 16) * 40, 40);
        #pragma unroll
        for (int im = 0; im < 4; ++im) {
            v16h a = frag_a(sA + (wm * 64 + im * 16) * 40, 40);
            acc[im][0] = wmma(a, b0, acc[im][0]);
            acc[im][1] = wmma(a, b1, acc[im][1]);
        }
        __syncthreads();
    }

    const int g = lane >> 4, ln = lane & 15;
    #pragma unroll
    for (int im = 0; im < 4; ++im)
        #pragma unroll
        for (int in = 0; in < 2; ++in)
            #pragma unroll
            for (int r = 0; r < 8; ++r) {
                int m = m0 + wm * 64 + im * 16 + r + 8 * g;
                int n = n0 + wn * 32 + in * 16 + ln;
                float v = acc[im][in][r] + bias[n];
                int b = m >> 11, s = m & 2047;
                int which = n >> 10, e = n & 1023, h = e >> 6, d = e & 63;
                size_t dst = ((size_t)((b * 16 + h) * 2048 + s)) * 64 + d;
                if (which == 0)      Q[dst] = (_Float16)(v * 0.125f); // 1/sqrt(64)
                else if (which == 1) K[dst] = (_Float16)v;
                else                 V[dst] = (_Float16)v;
            }
}

// ---------------------------------------------------------------------------
// Kernel 2: scores[bh,i,j] = Q[bh,i,:].K[bh,j,:] (Q pre-scaled), causal mask.
// 128x128 tiles; fully-masked tiles fill MASK_VALUE (NT). Q/K staged via
// async global->LDS copies.
// ---------------------------------------------------------------------------
__global__ __launch_bounds__(256)
void scores_kernel(const _Float16* __restrict__ Q, const _Float16* __restrict__ K,
                   float* __restrict__ attn) {
    const int tid = threadIdx.x;
    const int j0 = blockIdx.x * 128, i0 = blockIdx.y * 128;
    const int bh = blockIdx.z;

    if (j0 > i0 + 127) {              // tile entirely above diagonal: all masked
        const v4f mv = {MASK_VALUE, MASK_VALUE, MASK_VALUE, MASK_VALUE};
        #pragma unroll
        for (int it = 0; it < 16; ++it) {
            int idx = tid + it * 256;                  // 4096 float4's
            int r = idx >> 5, c4 = (idx & 31) * 4;
            __builtin_nontemporal_store(
                mv, (v4f*)(attn + ((size_t)bh * 2048 + i0 + r) * 2048 + j0 + c4));
        }
        return;
    }

    __shared__ __align__(16) _Float16 sQ[128 * 72];
    __shared__ __align__(16) _Float16 sK[128 * 72];
    const _Float16* Qb = Q + ((size_t)bh * 2048 + i0) * 64;
    const _Float16* Kb = K + ((size_t)bh * 2048 + j0) * 64;
    #pragma unroll
    for (int it = 0; it < 4; ++it) {                   // 1024 x 16B per matrix
        int idx = tid + it * 256;
        int r = idx >> 3, c8 = (idx & 7) * 8;
        cp16_g2l(Qb + r * 64 + c8, sQ + r * 72 + c8);
        cp16_g2l(Kb + r * 64 + c8, sK + r * 72 + c8);
    }
    async_join();
    __syncthreads();

    const int wave = tid >> 5, lane = tid & 31;
    const int wm = wave >> 2, wn = wave & 3;
    v8f acc[4][2];
    #pragma unroll
    for (int i = 0; i < 4; ++i)
        #pragma unroll
        for (int j = 0; j < 2; ++j) acc[i][j] = (v8f){0,0,0,0,0,0,0,0};

    #pragma unroll
    for (int kc = 0; kc < 64; kc += 32) {
        v16h b0 = frag_b(sK + (wn * 32 + 0)  * 72 + kc, 72);
        v16h b1 = frag_b(sK + (wn * 32 + 16) * 72 + kc, 72);
        #pragma unroll
        for (int im = 0; im < 4; ++im) {
            v16h a = frag_a(sQ + (wm * 64 + im * 16) * 72 + kc, 72);
            acc[im][0] = wmma(a, b0, acc[im][0]);
            acc[im][1] = wmma(a, b1, acc[im][1]);
        }
    }

    const int g = lane >> 4, ln = lane & 15;
    #pragma unroll
    for (int im = 0; im < 4; ++im)
        #pragma unroll
        for (int in = 0; in < 2; ++in)
            #pragma unroll
            for (int r = 0; r < 8; ++r) {
                int i = i0 + wm * 64 + im * 16 + r + 8 * g;
                int j = j0 + wn * 32 + in * 16 + ln;
                float v = acc[im][in][r];
                if (j > i) v = MASK_VALUE;
                __builtin_nontemporal_store(
                    v, attn + ((size_t)bh * 2048 + i) * 2048 + j);
            }
}

// ---------------------------------------------------------------------------
// Kernel 3: row softmax over 2048 elements, in place, NT traffic.
// ---------------------------------------------------------------------------
__global__ __launch_bounds__(256)
void softmax_kernel(float* __restrict__ attn) {
    const int tid = threadIdx.x;
    float* p = attn + (size_t)blockIdx.x * 2048;
    __shared__ float red[256];

    v4f c0 = __builtin_nontemporal_load((const v4f*)(p + tid * 4));
    v4f c1 = __builtin_nontemporal_load((const v4f*)(p + 1024 + tid * 4));
    float vals[8];
    #pragma unroll
    for (int k = 0; k < 4; ++k) { vals[k] = c0[k]; vals[k + 4] = c1[k]; }

    float mx = -3.0e38f;
    #pragma unroll
    for (int k = 0; k < 8; ++k) mx = fmaxf(mx, vals[k]);
    red[tid] = mx; __syncthreads();
    #pragma unroll
    for (int s = 128; s > 0; s >>= 1) {
        if (tid < s) red[tid] = fmaxf(red[tid], red[tid + s]);
        __syncthreads();
    }
    mx = red[0]; __syncthreads();

    float sum = 0.0f;
    #pragma unroll
    for (int k = 0; k < 8; ++k) { vals[k] = __expf(vals[k] - mx); sum += vals[k]; }
    red[tid] = sum; __syncthreads();
    #pragma unroll
    for (int s = 128; s > 0; s >>= 1) {
        if (tid < s) red[tid] += red[tid + s];
        __syncthreads();
    }
    float inv = 1.0f / red[0];
    #pragma unroll
    for (int k = 0; k < 4; ++k) { c0[k] = vals[k] * inv; c1[k] = vals[k + 4] * inv; }
    __builtin_nontemporal_store(c0, (v4f*)(p + tid * 4));
    __builtin_nontemporal_store(c1, (v4f*)(p + 1024 + tid * 4));
}

// ---------------------------------------------------------------------------
// Kernel 4: ctx[b,s, h*64+d] = sum_k attn[bh,s,k] * V[bh,k,d].
// M tile 128 x N 64, k-loop truncated at causal boundary. 8 waves (4x2).
// ---------------------------------------------------------------------------
__global__ __launch_bounds__(256)
void pv_gemm(const float* __restrict__ attn, const _Float16* __restrict__ V,
             float* __restrict__ ctx) {
    __shared__ __align__(16) _Float16 sP[128 * 40];
    __shared__ __align__(16) _Float16 sV[64 * 40];
    const int tid = threadIdx.x;
    const int i0 = blockIdx.x * 128;
    const int bh = blockIdx.y;
    const int wave = tid >> 5, lane = tid & 31;
    const int wm = wave >> 1, wn = wave & 1;

    v8f acc[2][2];
    #pragma unroll
    for (int i = 0; i < 2; ++i)
        #pragma unroll
        for (int j = 0; j < 2; ++j) acc[i][j] = (v8f){0,0,0,0,0,0,0,0};

    const int kmax = i0 + 128;        // beyond diagonal block the weights are 0
    for (int k0 = 0; k0 < kmax; k0 += 32) {
        v4f tp[4];
        #pragma unroll
        for (int it = 0; it < 4; ++it) {               // batch P loads (NT)
            int idx = tid + it * 256;
            int r = idx >> 3, c4 = (idx & 7) * 4;
            tp[it] = __builtin_nontemporal_load(
                (const v4f*)(attn + ((size_t)bh * 2048 + i0 + r) * 2048 + k0 + c4));
        }
        const int kk = tid >> 3, nn8 = (tid & 7) * 8;  // V: 256 x v8h = 2048 halfs
        v8h tv = *(const v8h*)(V + ((size_t)bh * 2048 + k0 + kk) * 64 + nn8);
        #pragma unroll
        for (int it = 0; it < 4; ++it) {               // convert+store P
            int idx = tid + it * 256;
            int r = idx >> 3, c4 = (idx & 7) * 4;
            *(v4h*)(sP + r * 40 + c4) = cvt4(tp[it]);
        }
        #pragma unroll
        for (int j = 0; j < 8; ++j) sV[(nn8 + j) * 40 + kk] = tv[j];
        __syncthreads();
        v16h b0 = frag_b(sV + (wn * 32 + 0)  * 40, 40);
        v16h b1 = frag_b(sV + (wn * 32 + 16) * 40, 40);
        #pragma unroll
        for (int im = 0; im < 2; ++im) {
            v16h a = frag_a(sP + (wm * 32 + im * 16) * 40, 40);
            acc[im][0] = wmma(a, b0, acc[im][0]);
            acc[im][1] = wmma(a, b1, acc[im][1]);
        }
        __syncthreads();
    }

    const int g = lane >> 4, ln = lane & 15;
    const int b = bh >> 4, h = bh & 15;
    #pragma unroll
    for (int im = 0; im < 2; ++im)
        #pragma unroll
        for (int in = 0; in < 2; ++in)
            #pragma unroll
            for (int r = 0; r < 8; ++r) {
                int s = i0 + wm * 32 + im * 16 + r + 8 * g;
                int d = wn * 32 + in * 16 + ln;
                ctx[((size_t)(b * 2048 + s)) * 1024 + h * 64 + d] = acc[im][in][r];
            }
}

// ---------------------------------------------------------------------------
// Kernel 5: out = ctx @ w_out + b_out.  M=8192, K=1024, N=1024.
// ---------------------------------------------------------------------------
__global__ __launch_bounds__(256)
void out_gemm(const float* __restrict__ ctx, const float* __restrict__ w,
              const float* __restrict__ bias, float* __restrict__ out) {
    __shared__ __align__(16) _Float16 sA[128 * 40];
    __shared__ __align__(16) _Float16 sB[128 * 40];
    const int tid = threadIdx.x;
    const int m0 = blockIdx.y * 128, n0 = blockIdx.x * 128;
    const int wave = tid >> 5, lane = tid & 31;
    const int wm = wave >> 2, wn = wave & 3;

    v8f acc[4][2];
    #pragma unroll
    for (int i = 0; i < 4; ++i)
        #pragma unroll
        for (int j = 0; j < 2; ++j) acc[i][j] = (v8f){0,0,0,0,0,0,0,0};

    for (int k0 = 0; k0 < 1024; k0 += 32) {
        v4f ta[4], tb[4];
        #pragma unroll
        for (int it = 0; it < 4; ++it) {               // batch A loads
            int idx = tid + it * 256;
            int r = idx >> 3, c4 = (idx & 7) * 4;
            ta[it] = *(const v4f*)(ctx + (size_t)(m0 + r) * 1024 + k0 + c4);
        }
        #pragma unroll
        for (int it = 0; it < 4; ++it) {               // batch B loads
            int idx = tid + it * 256;
            int kk = idx >> 5, nn4 = (idx & 31) * 4;
            tb[it] = *(const v4f*)(w + (size_t)(k0 + kk) * 1024 + n0 + nn4);
        }
        #pragma unroll
        for (int it = 0; it < 4; ++it) {               // convert+store A
            int idx = tid + it * 256;
            int r = idx >> 3, c4 = (idx & 7) * 4;
            *(v4h*)(sA + r * 40 + c4) = cvt4(ta[it]);
        }
        #pragma unroll
        for (int it = 0; it < 4; ++it) {               // convert+scatter B
            int idx = tid + it * 256;
            int kk = idx >> 5, nn4 = (idx & 31) * 4;
            v4h h = cvt4(tb[it]);
            #pragma unroll
            for (int j = 0; j < 4; ++j) sB[(nn4 + j) * 40 + kk] = h[j];
        }
        __syncthreads();
        v16h b0 = frag_b(sB + (wn * 32 + 0)  * 40, 40);
        v16h b1 = frag_b(sB + (wn * 32 + 16) * 40, 40);
        #pragma unroll
        for (int im = 0; im < 4; ++im) {
            v16h a = frag_a(sA + (wm * 64 + im * 16) * 40, 40);
            acc[im][0] = wmma(a, b0, acc[im][0]);
            acc[im][1] = wmma(a, b1, acc[im][1]);
        }
        __syncthreads();
    }

    const int g = lane >> 4, ln = lane & 15;
    #pragma unroll
    for (int im = 0; im < 4; ++im)
        #pragma unroll
        for (int in = 0; in < 2; ++in)
            #pragma unroll
            for (int r = 0; r < 8; ++r) {
                int m = m0 + wm * 64 + im * 16 + r + 8 * g;
                int n = n0 + wn * 32 + in * 16 + ln;
                out[(size_t)m * 1024 + n] = acc[im][in][r] + bias[n];
            }
}

// ---------------------------------------------------------------------------
extern "C" void kernel_launch(void* const* d_in, const int* in_sizes, int n_in,
                              void* d_out, int out_size, void* d_ws, size_t ws_size,
                              hipStream_t stream) {
    const float* x     = (const float*)d_in[0];   // [4,2048,1024]
    const float* w_qkv = (const float*)d_in[1];   // [1024,3072]
    const float* b_qkv = (const float*)d_in[2];   // [3072]
    const float* w_out = (const float*)d_in[3];   // [1024,1024]
    const float* b_out = (const float*)d_in[4];   // [1024]

    float* out  = (float*)d_out;                       // [4,2048,1024]
    float* attn = out + (size_t)4 * 2048 * 1024;       // [4,16,2048,2048]

    const size_t nQKV = (size_t)4 * 16 * 2048 * 64;    // 8,388,608 elems/tensor
    _Float16* Q = (_Float16*)d_ws;
    _Float16* K = Q + nQKV;
    _Float16* V = K + nQKV;
    float*  ctx = (float*)(V + nQKV);                  // [4,2048,1024] fp32

    qkv_gemm     <<<dim3(24, 64),       256, 0, stream>>>(x, w_qkv, b_qkv, Q, K, V);
    scores_kernel<<<dim3(16, 16, 64),   256, 0, stream>>>(Q, K, attn);
    softmax_kernel<<<dim3(4 * 16 * 2048), 256, 0, stream>>>(attn);
    pv_gemm      <<<dim3(16, 64),       256, 0, stream>>>(attn, V, ctx);
    out_gemm     <<<dim3(8, 64),        256, 0, stream>>>(ctx, w_out, b_out, out);
}
